// MLAAttention_10969346474276
// MI455X (gfx1250) — matrix operand order repaired and checked by reading, not compiled, not run on hardware
//
#include <hip/hip_runtime.h>
#include <stdint.h>

// ---------------------------------------------------------------------------
// CDNA5 (gfx1250) MLA attention: bf16 WMMA everywhere, flash-attention fusion,
// double-buffered LDS staging, async direct-to-LDS copies for untransposed
// tiles (GLOBAL_LOAD_ASYNC_TO_LDS_B128, ASYNCcnt-tracked).
// ---------------------------------------------------------------------------

typedef __attribute__((ext_vector_type(8)))  __bf16 v8bf;
typedef __attribute__((ext_vector_type(16))) __bf16 v16bf;
typedef __attribute__((ext_vector_type(8)))  float  v8f;

__device__ __forceinline__ v16bf cat8(v8bf lo, v8bf hi) {
  return __builtin_shufflevector(lo, hi, 0,1,2,3,4,5,6,7,8,9,10,11,12,13,14,15);
}

__device__ __forceinline__ v8f wmma_bf16f32(v16bf a, v16bf b, v8f c) {
  // D = A(16x32 bf16) * B(32x16 bf16) + C(16x16 f32)
  return __builtin_amdgcn_wmma_f32_16x16x32_bf16(false, a, false, b,
                                                 (short)0, c, false, false);
}

// Async global->LDS 16B copy (gfx1250). ldsoff = wave-relative LDS byte addr
// (low 32 bits of a generic shared pointer). Tracked by ASYNCcnt.
__device__ __forceinline__ void async_ld16(uint32_t ldsoff, const void* g) {
  asm volatile("global_load_async_to_lds_b128 %0, %1, off"
               :: "v"(ldsoff), "v"(g) : "memory");
}
__device__ __forceinline__ void wait_async0() {
  asm volatile("s_wait_asynccnt 0x0" ::: "memory");
}
__device__ __forceinline__ uint32_t lds_addr(const void* p) {
  return (uint32_t)(uintptr_t)p;   // generic shared ptr: addr[31:0] = LDS offset
}

// ---------------------------------------------------------------------------
// Generic GEMM: C[M,N] = A[M,K] @ B[K,N]; bf16 in/out, fp32 accumulate.
// 128 threads (4 waves). Workgroup tile 64x64; wave tile 16x64 (4 WMMA accs).
// Double-buffered LDS: stage K-step s+1 (A via async-to-LDS, B via transpose
// store) while WMMAs run on step s. All 4 B-fragments preloaded so the four
// v_wmma issue back-to-back.
// ---------------------------------------------------------------------------
__global__ __launch_bounds__(128) void gemm_bf16(
    const __bf16* __restrict__ A, const __bf16* __restrict__ B,
    __bf16* __restrict__ C, int M, int N, int K)
{
  __shared__ __align__(16) __bf16 sA [2][64][40];   // [buf][m][k], padded rows
  __shared__ __align__(16) __bf16 sBt[2][64][40];   // [buf][n][k], transposed

  const int tid  = threadIdx.x;
  const int m0   = blockIdx.y * 64;
  const int n0   = blockIdx.x * 64;
  const int w    = tid >> 5;
  const int lane = tid & 31;
  const int l16  = lane & 15;
  const int hi   = lane >> 4;

  const int arow = tid >> 1, ahalf = tid & 1;   // A: 64 rows x 32 k, 16 elt/thr
  const int bkk  = tid >> 2, bseg  = tid & 3;   // B: 32 k x 64 n, 16 elt/thr

  v8f acc[4];
  #pragma unroll
  for (int n = 0; n < 4; ++n)
    #pragma unroll
    for (int e = 0; e < 8; ++e) acc[n][e] = 0.f;

  const int nsteps = K >> 5;

  auto stage = [&](int buf, int k0) {
    // A tile: no transpose needed -> async direct to LDS (2 x 16B per thread)
    const __bf16* ap = A + (size_t)(m0 + arow) * K + k0 + ahalf * 16;
    const uint32_t la = lds_addr(&sA[buf][arow][ahalf * 16]);
    async_ld16(la,      ap);
    async_ld16(la + 16, ap + 8);
    // B tile: transpose on store (VGPR round-trip required)
    const __bf16* bp = B + (size_t)(k0 + bkk) * N + n0 + bseg * 16;
    v8bf b0 = *(const v8bf*)(bp);
    v8bf b1 = *(const v8bf*)(bp + 8);
    #pragma unroll
    for (int i = 0; i < 8; ++i) {
      sBt[buf][bseg * 16 + i    ][bkk] = b0[i];
      sBt[buf][bseg * 16 + 8 + i][bkk] = b1[i];
    }
  };

  auto compute = [&](int buf) {
    // A fragment: lane row = w*16 + l16; runs at k={hi*8, 16+hi*8}
    const __bf16* ar = &sA[buf][w * 16 + l16][0];
    v16bf af = cat8(*(const v8bf*)&ar[hi * 8],
                    *(const v8bf*)&ar[16 + hi * 8]);
    // Preload all 4 B fragments (distinct regs -> DS loads pipeline,
    // WMMAs issue back-to-back).
    v16bf bm[4];
    #pragma unroll
    for (int n = 0; n < 4; ++n) {
      const __bf16* br = &sBt[buf][n * 16 + l16][0];
      bm[n] = cat8(*(const v8bf*)&br[hi * 8],
                   *(const v8bf*)&br[16 + hi * 8]);
    }
    #pragma unroll
    for (int n = 0; n < 4; ++n) acc[n] = wmma_bf16f32(af, bm[n], acc[n]);
  };

  stage(0, 0);
  wait_async0();
  __syncthreads();
  for (int s = 1; s < nsteps; ++s) {
    const int buf = s & 1;
    stage(buf, s * 32);                 // overlaps with compute below
    if (s + 1 < nsteps)                 // warm L2 for the tile after next
      __builtin_prefetch(A + (size_t)(m0 + arow) * K + (s + 1) * 32, 0, 1);
    compute(buf ^ 1);
    wait_async0();
    __syncthreads();
  }
  compute((nsteps - 1) & 1);

  // C layout: element e of acc[n] -> row = hi*8+e, col = n*16+l16
  #pragma unroll
  for (int n = 0; n < 4; ++n)
    #pragma unroll
    for (int e = 0; e < 8; ++e) {
      const int row = m0 + w * 16 + hi * 8 + e;
      const int col = n0 + n * 16 + l16;
      C[(size_t)row * N + col] = (__bf16)acc[n][e];
    }
}

// ---------------------------------------------------------------------------
// RMSNorm over rows of length L (in-place, bf16 data, f32 scale).
// One wave per row; 8 rows per 256-thread block.
// ---------------------------------------------------------------------------
__global__ __launch_bounds__(256) void rmsnorm_rows(
    __bf16* __restrict__ X, const float* __restrict__ scale, int L)
{
  const int row  = blockIdx.x * 8 + (threadIdx.x >> 5);
  const int lane = threadIdx.x & 31;
  __bf16* r = X + (size_t)row * L;
  float ss = 0.f;
  for (int i = lane; i < L; i += 32) { float v = (float)r[i]; ss += v * v; }
  #pragma unroll
  for (int mk = 16; mk >= 1; mk >>= 1) ss += __shfl_xor(ss, mk, 32);
  const float inv = rsqrtf(ss / (float)L + 1e-6f);
  for (int i = lane; i < L; i += 32)
    r[i] = (__bf16)((float)r[i] * inv * scale[i]);
}

// ---------------------------------------------------------------------------
// RoPE in-place on [T, nheads, 64]; one thread per (t, h, pair d<32).
// ---------------------------------------------------------------------------
__global__ __launch_bounds__(256) void rope_inplace(
    __bf16* __restrict__ X, const __bf16* __restrict__ sinp,
    const __bf16* __restrict__ cosp, int nheads, int total)
{
  const int idx = blockIdx.x * 256 + threadIdx.x;
  if (idx >= total) return;
  const int d = idx & 31;
  const int h = (idx >> 5) % nheads;
  const int t = idx / (32 * nheads);
  __bf16* p = X + ((size_t)t * nheads + h) * 64;
  const float x0 = (float)p[d], x1 = (float)p[d + 32];
  const float c0 = (float)cosp[t * 64 + d], c1 = (float)cosp[t * 64 + d + 32];
  const float s0 = (float)sinp[t * 64 + d], s1 = (float)sinp[t * 64 + d + 32];
  p[d]      = (__bf16)(x0 * c0 - x1 * s0);
  p[d + 32] = (__bf16)(x1 * c1 + x0 * s1);
}

// ---------------------------------------------------------------------------
// Flash attention (causal, GQA 16q/4kv heads, head_dim=64, scale=1/8).
// Grid: (T/64 query blocks, 16 heads); 128 threads = 4 waves.
// K staged row-major via async-to-LDS; V staged transposed via VGPRs;
// P bounced through per-wave LDS (C-layout -> A-layout) between WMMAs.
// ---------------------------------------------------------------------------
__global__ __launch_bounds__(128) void flash_attn(
    const __bf16* __restrict__ Q, const __bf16* __restrict__ Kb,
    const __bf16* __restrict__ Vb, __bf16* __restrict__ O)
{
  __shared__ __align__(16) __bf16 sK [64][72];      // [key][dim]
  __shared__ __align__(16) __bf16 sVt[64][72];      // [dim][key]
  __shared__ __align__(16) __bf16 sP [4][16][72];   // per-wave P tile

  const int qb  = blockIdx.x;
  const int h   = blockIdx.y;
  const int kvh = h >> 2;                            // GQA repeat = 4
  const int tid = threadIdx.x;
  const int w = tid >> 5, lane = tid & 31, l16 = lane & 15, hi = lane >> 4;

  // Q fragments (16x64 -> two 16x32 A-frags), loaded straight from global.
  v16bf qf[2];
  {
    const __bf16* qp = Q + (size_t)(qb * 64 + w * 16 + l16) * 1024 + h * 64;
    #pragma unroll
    for (int c = 0; c < 2; ++c)
      qf[c] = cat8(*(const v8bf*)(qp + c * 32 + hi * 8),
                   *(const v8bf*)(qp + c * 32 + 16 + hi * 8));
  }

  v8f oacc[4];
  float m_i[8], l_i[8];
  #pragma unroll
  for (int n = 0; n < 4; ++n)
    #pragma unroll
    for (int e = 0; e < 8; ++e) oacc[n][e] = 0.f;
  #pragma unroll
  for (int e = 0; e < 8; ++e) { m_i[e] = -3.0e38f; l_i[e] = 0.f; }

  const int srow = tid >> 1, sseg = tid & 1;         // staging split

  for (int kb = 0; kb <= qb; ++kb) {
    __syncthreads();
    {
      const size_t base = (size_t)(kb * 64 + srow) * 256 + kvh * 64 + sseg * 32;
      // K rows: no transpose -> async direct to LDS (4 x 16B per thread)
      const __bf16* kp = Kb + base;
      #pragma unroll
      for (int j = 0; j < 4; ++j)
        async_ld16(lds_addr(&sK[srow][sseg * 32 + j * 8]), kp + j * 8);
      // V rows: transpose on store
      const __bf16* vp = Vb + base;
      #pragma unroll
      for (int j = 0; j < 4; ++j) {
        v8bf vv = *(const v8bf*)(vp + j * 8);
        #pragma unroll
        for (int i = 0; i < 8; ++i)
          sVt[sseg * 32 + j * 8 + i][srow] = vv[i];
      }
    }
    wait_async0();
    __syncthreads();

    // ---- S = Q @ K^T  (16x64 scores per wave, 8 WMMAs) ----
    v8f s[4];
    #pragma unroll
    for (int n = 0; n < 4; ++n)
      #pragma unroll
      for (int e = 0; e < 8; ++e) s[n][e] = 0.f;
    #pragma unroll
    for (int c = 0; c < 2; ++c) {
      v16bf bm[4];
      #pragma unroll
      for (int n = 0; n < 4; ++n) {
        const __bf16* br = &sK[n * 16 + l16][c * 32];  // key = n*16+l16
        bm[n] = cat8(*(const v8bf*)&br[hi * 8],
                     *(const v8bf*)&br[16 + hi * 8]);
      }
      #pragma unroll
      for (int n = 0; n < 4; ++n) s[n] = wmma_bf16f32(qf[c], bm[n], s[n]);
    }

    // ---- scale + causal mask + online softmax ----
    const bool diag = (kb == qb);
    float rmax[8];
    #pragma unroll
    for (int e = 0; e < 8; ++e) rmax[e] = -3.0e38f;
    #pragma unroll
    for (int n = 0; n < 4; ++n)
      #pragma unroll
      for (int e = 0; e < 8; ++e) {
        float v = s[n][e] * 0.125f;                    // 1/sqrt(64)
        if (diag && (n * 16 + l16) > (w * 16 + hi * 8 + e)) v = -3.0e38f;
        s[n][e] = v;
        rmax[e] = fmaxf(rmax[e], v);
      }
    #pragma unroll
    for (int mk = 1; mk < 16; mk <<= 1)
      #pragma unroll
      for (int e = 0; e < 8; ++e)
        rmax[e] = fmaxf(rmax[e], __shfl_xor(rmax[e], mk, 32));

    float rsum[8];
    #pragma unroll
    for (int e = 0; e < 8; ++e) {
      const float mnew = fmaxf(m_i[e], rmax[e]);
      const float corr = __expf(m_i[e] - mnew);
      m_i[e] = mnew;
      l_i[e] *= corr;
      #pragma unroll
      for (int n = 0; n < 4; ++n) oacc[n][e] *= corr;
      rsum[e] = 0.f;
    }
    #pragma unroll
    for (int n = 0; n < 4; ++n)
      #pragma unroll
      for (int e = 0; e < 8; ++e) {
        const float p = __expf(s[n][e] - m_i[e]);
        s[n][e] = p;
        rsum[e] += p;
      }
    #pragma unroll
    for (int mk = 1; mk < 16; mk <<= 1)
      #pragma unroll
      for (int e = 0; e < 8; ++e) rsum[e] += __shfl_xor(rsum[e], mk, 32);
    #pragma unroll
    for (int e = 0; e < 8; ++e) l_i[e] += rsum[e];

    // ---- bounce P through LDS (C-layout -> A-layout), bf16 like reference ----
    #pragma unroll
    for (int n = 0; n < 4; ++n)
      #pragma unroll
      for (int e = 0; e < 8; ++e)
        sP[w][hi * 8 + e][n * 16 + l16] = (__bf16)s[n][e];
    __syncthreads();

    // ---- O += P @ V  (8 WMMAs) ----
    #pragma unroll
    for (int c = 0; c < 2; ++c) {
      const __bf16* pr = &sP[w][l16][0];
      v16bf pf = cat8(*(const v8bf*)&pr[c * 32 + hi * 8],
                      *(const v8bf*)&pr[c * 32 + 16 + hi * 8]);
      v16bf bm[4];
      #pragma unroll
      for (int n = 0; n < 4; ++n) {
        const __bf16* br = &sVt[n * 16 + l16][c * 32]; // dim = n*16+l16
        bm[n] = cat8(*(const v8bf*)&br[hi * 8],
                     *(const v8bf*)&br[16 + hi * 8]);
      }
      #pragma unroll
      for (int n = 0; n < 4; ++n) oacc[n] = wmma_bf16f32(pf, bm[n], oacc[n]);
    }
  }

  // ---- normalize and write out [t, h*64 + d] ----
  #pragma unroll
  for (int n = 0; n < 4; ++n)
    #pragma unroll
    for (int e = 0; e < 8; ++e) {
      const int t = qb * 64 + w * 16 + hi * 8 + e;
      const int d = n * 16 + l16;
      O[(size_t)t * 1024 + h * 64 + d] = (__bf16)(oacc[n][e] / l_i[e]);
    }
}

// ---------------------------------------------------------------------------
// Host-side orchestration
// ---------------------------------------------------------------------------
extern "C" void kernel_launch(void* const* d_in, const int* in_sizes, int n_in,
                              void* d_out, int out_size, void* d_ws, size_t ws_size,
                              hipStream_t stream) {
  (void)in_sizes; (void)n_in; (void)out_size; (void)ws_size;
  const int T = 4096, D = 1024;
  const __bf16* x        = (const __bf16*)d_in[0];
  const __bf16* sinp     = (const __bf16*)d_in[1];
  const __bf16* cosp     = (const __bf16*)d_in[2];
  const __bf16* w_dq     = (const __bf16*)d_in[3];
  const float*  scale_q  = (const float*)d_in[4];
  const __bf16* w_uq     = (const __bf16*)d_in[5];
  const __bf16* w_dkv    = (const __bf16*)d_in[6];
  const float*  scale_kv = (const float*)d_in[7];
  const __bf16* w_uk     = (const __bf16*)d_in[8];
  const __bf16* w_uv     = (const __bf16*)d_in[9];
  const __bf16* w_o      = (const __bf16*)d_in[10];
  __bf16* out = (__bf16*)d_out;

  char* ws = (char*)d_ws;
  __bf16* q_lat  = (__bf16*)(ws);                      // 4096x512  (4 MB)
  __bf16* kv_lat = (__bf16*)(ws + (4u   << 20));       // 4096x256  (2 MB)
  __bf16* qbuf   = (__bf16*)(ws + (6u   << 20));       // 4096x1024 (8 MB)
  __bf16* kbuf   = (__bf16*)(ws + (14u  << 20));       // 4096x256  (2 MB)
  __bf16* vbuf   = (__bf16*)(ws + (16u  << 20));       // 4096x256  (2 MB)
  __bf16* abuf   = (__bf16*)(ws + (18u  << 20));       // 4096x1024 (8 MB)

  // 1) latent projections
  gemm_bf16<<<dim3(512 / 64, T / 64), 128, 0, stream>>>(x, w_dq,  q_lat,  T, 512, D);
  gemm_bf16<<<dim3(256 / 64, T / 64), 128, 0, stream>>>(x, w_dkv, kv_lat, T, 256, D);
  // 2) RMS norms (in place)
  rmsnorm_rows<<<T / 8, 256, 0, stream>>>(q_lat,  scale_q,  512);
  rmsnorm_rows<<<T / 8, 256, 0, stream>>>(kv_lat, scale_kv, 256);
  // 3) up-projections
  gemm_bf16<<<dim3(1024 / 64, T / 64), 128, 0, stream>>>(q_lat,  w_uq, qbuf, T, 1024, 512);
  gemm_bf16<<<dim3(256  / 64, T / 64), 128, 0, stream>>>(kv_lat, w_uk, kbuf, T, 256,  256);
  gemm_bf16<<<dim3(256  / 64, T / 64), 128, 0, stream>>>(kv_lat, w_uv, vbuf, T, 256,  256);
  // 4) RoPE on q (16 heads) and k (4 heads), in place
  rope_inplace<<<(T * 16 * 32) / 256, 256, 0, stream>>>(qbuf, sinp, cosp, 16, T * 16 * 32);
  rope_inplace<<<(T * 4  * 32) / 256, 256, 0, stream>>>(kbuf, sinp, cosp, 4,  T * 4 * 32);
  // 5) fused causal attention
  flash_attn<<<dim3(T / 64, 16), 128, 0, stream>>>(qbuf, kbuf, vbuf, abuf);
  // 6) output projection
  gemm_bf16<<<dim3(1024 / 64, T / 64), 128, 0, stream>>>(abuf, w_o, out, T, 1024, D);
}
